// CrossAttention_18897856103166
// MI455X (gfx1250) — compile-verified, matrix operand order checked by reading, not run on hardware
//
#include <hip/hip_runtime.h>
#include <cstdint>
#include <cstddef>

// ---------------------------------------------------------------------------
// MI455X (gfx1250, wave32) fused cross-attention.
//   ~51.6 GFLOP, ~60MB working set (L2-resident): compute-bound -> all GEMMs
//   on v_wmma_f32_16x16x32_f16, flash-attention fusion (never materialize the
//   537MB score matrix).  Attention inner loop computes S^T so softmax needs
//   only one 16-lane exchange per reduction, K/V tiles are double-buffered in
//   LDS via global_load_async_to_lds_b128 (ASYNCcnt), and P re-layout uses
//   packed ds_store_b128.
// ---------------------------------------------------------------------------

typedef __attribute__((ext_vector_type(16))) _Float16 v16h;
typedef __attribute__((ext_vector_type(8)))  _Float16 v8h;
typedef __attribute__((ext_vector_type(8)))  float    v8f;
typedef __attribute__((ext_vector_type(4)))  float    v4f;

union F16Frag { v16h v; v8h h[2]; };

__device__ __forceinline__ v8f wmma_f16(v16h a, v16h b, v8f c) {
  // D = A(16x32) * B(32x16) + C(16x16 f32)
  return __builtin_amdgcn_wmma_f32_16x16x32_f16(false, a, false, b, (short)0, c,
                                                false, false);
}

// Load one 16x32 f16 operand fragment (A rows, or B columns stored
// K-contiguous). Per ISA 7.12.2: lanes 0-15 hold row r, K {0..7,16..23};
// lanes 16-31 hold row r, K {8..15,24..31}.  Two 16B chunks per lane.
__device__ __forceinline__ v16h load_frag16(const _Float16* base, int stride,
                                            int lane) {
  const int r   = lane & 15;
  const int sel = (lane >> 4) & 1;
  const _Float16* p = base + (size_t)r * stride + sel * 8;
  F16Frag u;
  u.h[0] = *(const v8h*)(p);
  u.h[1] = *(const v8h*)(p + 16);
  return u.v;
}

// Same fragment, sourced from fp32 memory with inline f16 conversion.
__device__ __forceinline__ v16h load_frag16_f32(const float* base, int stride,
                                                int lane) {
  const int r   = lane & 15;
  const int sel = (lane >> 4) & 1;
  const float* p = base + (size_t)r * stride + sel * 8;
  v4f a0 = *(const v4f*)(p);
  v4f a1 = *(const v4f*)(p + 4);
  v4f b0 = *(const v4f*)(p + 16);
  v4f b1 = *(const v4f*)(p + 20);
  F16Frag u;
#pragma unroll
  for (int i = 0; i < 4; ++i) {
    u.h[0][i]     = (_Float16)a0[i];
    u.h[0][4 + i] = (_Float16)a1[i];
    u.h[1][i]     = (_Float16)b0[i];
    u.h[1][4 + i] = (_Float16)b1[i];
  }
  return u.v;
}

// CDNA5 async memory->LDS copy, 16B per lane, tracked by ASYNCcnt.
__device__ __forceinline__ void async_copy_b128(uint32_t lds_byte_addr,
                                                const void* gptr) {
  asm volatile("global_load_async_to_lds_b128 %0, %1, off"
               :
               : "v"(lds_byte_addr), "v"((uint64_t)(uintptr_t)gptr)
               : "memory");
}
__device__ __forceinline__ void wait_async0() {
  asm volatile("s_wait_asynccnt 0x0" ::: "memory");
}

// ---------------------------------------------------------------------------
// Kernel 0: W[512][512] fp32 row-major -> WT[c][k] f16 (B-operand columns
// become K-contiguous).
// ---------------------------------------------------------------------------
__global__ void transpose_w_kernel(const float* __restrict__ W,
                                   _Float16* __restrict__ WT) {
  int idx = blockIdx.x * 256 + threadIdx.x;     // 512*512 elements
  int k = idx >> 9, c = idx & 511;
  WT[(size_t)c * 512 + k] = (_Float16)W[idx];
}

// ---------------------------------------------------------------------------
// Kernel 1: projection GEMM.  X[8192,512] fp32 @ WT(f16, col-major) ->
// f16 output.  One 16x64 tile per wave, 8 waves/WG -> block covers 128x64.
// VTRANS=0: Out[row][col]  (Q, K:  [B,N,512])
// VTRANS=1: Out stored per-head transposed: [B,H,64,M]  (V for the PV GEMM)
// ---------------------------------------------------------------------------
template <int VTRANS>
__global__ __launch_bounds__(256) void proj_gemm_kernel(
    const float* __restrict__ X, const _Float16* __restrict__ WT,
    _Float16* __restrict__ Out) {
  const int lane = threadIdx.x & 31;
  const int wave = threadIdx.x >> 5;
  const int row0 = blockIdx.x * 128 + wave * 16;  // [0, 8192)
  const int col0 = blockIdx.y * 64;               // [0, 512)

  v8f acc[4] = {};
  const float* xrow = X + (size_t)row0 * 512;
#pragma unroll 2
  for (int k0 = 0; k0 < 512; k0 += 32) {
    v16h a = load_frag16_f32(xrow + k0, 512, lane);
#pragma unroll
    for (int j = 0; j < 4; ++j) {
      v16h b = load_frag16(WT + (size_t)(col0 + j * 16) * 512 + k0, 512, lane);
      acc[j] = wmma_f16(a, b, acc[j]);
    }
  }

  const int cl = lane & 15;
  const int rb = (lane >> 4) * 8;   // C layout: VGPR r -> row r (lo) / r+8 (hi)
#pragma unroll
  for (int j = 0; j < 4; ++j) {
    const int c = col0 + j * 16 + cl;
#pragma unroll
    for (int r = 0; r < 8; ++r) {
      const int row = row0 + rb + r;
      _Float16 hv = (_Float16)acc[j][r];
      if (VTRANS) {
        int b = row >> 11, m = row & 2047;   // N = 2048
        int h = c >> 6, d = c & 63;
        Out[((size_t)((b * 8 + h) * 64 + d)) * 2048 + m] = hv;
      } else {
        Out[(size_t)row * 512 + c] = hv;
      }
    }
  }
}

// ---------------------------------------------------------------------------
// Kernel 2: flash attention.  Block = (b, h, 128 query rows); 8 waves of 16
// query columns each.  Computes S^T = K*Q^T so each lane owns one query's
// softmax stats (one xor-16 exchange per reduction).  K/V 64x64 tiles are
// double-buffered in LDS with async global->LDS copies.
// ---------------------------------------------------------------------------
__global__ __launch_bounds__(256) void attn_kernel(
    const _Float16* __restrict__ Qf,   // [B, N, 512] f16
    const _Float16* __restrict__ Kf,   // [B, M, 512] f16
    const _Float16* __restrict__ Vt,   // [B, H, 64, M] f16
    _Float16* __restrict__ Of) {       // [B, N, 512] f16
  __shared__ _Float16 sK[2][64 * 64];   // [m][d]
  __shared__ _Float16 sV[2][64 * 64];   // [d][m]
  __shared__ _Float16 sP[8][16 * 64];   // per-wave P staging [n][m]
  __shared__ __align__(64) float sBcast[8][16];  // per-wave row broadcast

  const int lane = threadIdx.x & 31;
  const int wave = threadIdx.x >> 5;
  const int b = blockIdx.z, h = blockIdx.y;
  const int n0 = blockIdx.x * 128 + wave * 16;
  const int t = threadIdx.x;

  // Q as the B operand (columns = queries), pre-scaled by 1/sqrt(64)=0.125.
  v16h bq[2];
  {
    const _Float16* qbase = Qf + ((size_t)(b * 2048 + n0)) * 512 + h * 64;
    bq[0] = load_frag16(qbase, 512, lane);
    bq[1] = load_frag16(qbase + 32, 512, lane);
#pragma unroll
    for (int i = 0; i < 16; ++i) {
      bq[0][i] *= (_Float16)0.125f;
      bq[1][i] *= (_Float16)0.125f;
    }
  }

  v8f o[4] = {};
  float m_n = -1e30f, l_n = 0.f;   // per-lane: stats for query n = lane&15

  const _Float16* kbase = Kf + (size_t)b * 2048 * 512 + h * 64;
  const _Float16* vbase = Vt + (size_t)((b * 8 + h) * 64) * 2048;

  // Cooperative async tile fetch: 2 x (64x64 f16) = 2 x 512 16B chunks,
  // 256 threads x 2 chunks each per tile.
  auto issue_tile = [&](int m0, int buf) {
#pragma unroll
    for (int i = 0; i < 2; ++i) {
      int idx = t + i * 256;
      int rr = idx >> 3, cc = (idx & 7) * 8;
      async_copy_b128((uint32_t)(uintptr_t)&sK[buf][rr * 64 + cc],
                      kbase + (size_t)(m0 + rr) * 512 + cc);
      async_copy_b128((uint32_t)(uintptr_t)&sV[buf][rr * 64 + cc],
                      vbase + (size_t)rr * 2048 + m0 + cc);
    }
  };

  issue_tile(0, 0);

  const int rb = (lane >> 4) * 8;
  const int n = lane & 15;

  for (int step = 0; step < 32; ++step) {
    const int buf = step & 1;
    wait_async0();        // this wave's tile writes have landed
    __syncthreads();      // everyone's landed; prior buffer reads complete
    if (step + 1 < 32) issue_tile((step + 1) * 64, buf ^ 1);

    const _Float16* tK = sK[buf];
    const _Float16* tV = sV[buf];

    // S^T = K * Q^T : A rows = key positions m, B columns = queries n.
    v8f st[4] = {};
#pragma unroll
    for (int j = 0; j < 4; ++j) {
#pragma unroll
      for (int kk = 0; kk < 2; ++kk) {
        v16h ak = load_frag16(tK + (j * 16) * 64 + kk * 32, 64, lane);
        st[j] = wmma_f16(ak, bq[kk], st[j]);
      }
    }

    // Online softmax: lane holds 32 of query n's 64 scores; the other 32 are
    // in lane^16.  One exchange for max, one for sum.
    float mloc = -1e30f;
#pragma unroll
    for (int j = 0; j < 4; ++j)
#pragma unroll
      for (int r = 0; r < 8; ++r) mloc = fmaxf(mloc, st[j][r]);
    mloc = fmaxf(mloc, __shfl_xor(mloc, 16, 32));
    float mnew = fmaxf(m_n, mloc);
    float corr = __expf(m_n - mnew);
    float rs = 0.f;
#pragma unroll
    for (int j = 0; j < 4; ++j)
#pragma unroll
      for (int r = 0; r < 8; ++r) {
        float e = __expf(st[j][r] - mnew);
        st[j][r] = e;
        rs += e;
      }
    rs += __shfl_xor(rs, 16, 32);
    m_n = mnew;
    l_n = l_n * corr + rs;

    // Broadcast corr into the O C-layout (rows indexed per-VGPR): via 16
    // floats of per-wave LDS.
    if (lane < 16) sBcast[wave][lane] = corr;
    v4f c0 = *(const v4f*)&sBcast[wave][rb];
    v4f c1 = *(const v4f*)&sBcast[wave][rb + 4];
#pragma unroll
    for (int f = 0; f < 4; ++f)
#pragma unroll
      for (int r = 0; r < 4; ++r) {
        o[f][r]     *= c0[r];
        o[f][4 + r] *= c1[r];
      }

    // P^T C-layout -> packed f16 -> per-wave LDS [n][m]: each lane's 8
    // fragment values are m-consecutive -> one ds_store_b128 per sub-tile.
    _Float16* pw = sP[wave];
#pragma unroll
    for (int j = 0; j < 4; ++j) {
      v8h ph;
#pragma unroll
      for (int r = 0; r < 8; ++r) ph[r] = (_Float16)st[j][r];
      *(v8h*)&pw[n * 64 + j * 16 + rb] = ph;
    }

    // O += P * V  (A rows = queries from sP, B columns = d from sV)
    v16h ap0 = load_frag16(pw, 64, lane);
    v16h ap1 = load_frag16(pw + 32, 64, lane);
#pragma unroll
    for (int f = 0; f < 4; ++f) {
      v16h bv0 = load_frag16(tV + (f * 16) * 64, 64, lane);
      v16h bv1 = load_frag16(tV + (f * 16) * 64 + 32, 64, lane);
      o[f] = wmma_f16(ap0, bv0, o[f]);
      o[f] = wmma_f16(ap1, bv1, o[f]);
    }
  }

  // Normalize (broadcast 1/l the same way) and write the O tile.
  {
    float inv = 1.f / l_n;
    if (lane < 16) sBcast[wave][lane] = inv;
    v4f c0 = *(const v4f*)&sBcast[wave][rb];
    v4f c1 = *(const v4f*)&sBcast[wave][rb + 4];
#pragma unroll
    for (int f = 0; f < 4; ++f)
#pragma unroll
      for (int r = 0; r < 4; ++r) {
        o[f][r]     *= c0[r];
        o[f][4 + r] *= c1[r];
      }
  }
  _Float16* obase = Of + ((size_t)(b * 2048 + n0)) * 512 + h * 64;
#pragma unroll
  for (int f = 0; f < 4; ++f)
#pragma unroll
    for (int r = 0; r < 8; ++r)
      obase[(size_t)(rb + r) * 512 + f * 16 + n] = (_Float16)o[f][r];
}

// ---------------------------------------------------------------------------
// Kernel 3: output projection.  O(f16) @ WoT(f16) + bo -> fp32 result.
// ---------------------------------------------------------------------------
__global__ __launch_bounds__(256) void out_gemm_kernel(
    const _Float16* __restrict__ Of, const _Float16* __restrict__ WoT,
    const float* __restrict__ bo, float* __restrict__ Y) {
  const int lane = threadIdx.x & 31;
  const int wave = threadIdx.x >> 5;
  const int row0 = blockIdx.x * 128 + wave * 16;
  const int col0 = blockIdx.y * 64;

  v8f acc[4] = {};
  const _Float16* arow = Of + (size_t)row0 * 512;
#pragma unroll 2
  for (int k0 = 0; k0 < 512; k0 += 32) {
    v16h a = load_frag16(arow + k0, 512, lane);
#pragma unroll
    for (int j = 0; j < 4; ++j) {
      v16h b = load_frag16(WoT + (size_t)(col0 + j * 16) * 512 + k0, 512, lane);
      acc[j] = wmma_f16(a, b, acc[j]);
    }
  }

  const int cl = lane & 15;
  const int rb = (lane >> 4) * 8;
#pragma unroll
  for (int j = 0; j < 4; ++j) {
    const int c = col0 + j * 16 + cl;
    const float bias = bo[c];
#pragma unroll
    for (int r = 0; r < 8; ++r)
      Y[(size_t)(row0 + rb + r) * 512 + c] = acc[j][r] + bias;
  }
}

// ---------------------------------------------------------------------------
extern "C" void kernel_launch(void* const* d_in, const int* in_sizes, int n_in,
                              void* d_out, int out_size, void* d_ws,
                              size_t ws_size, hipStream_t stream) {
  const float* x   = (const float*)d_in[0];   // [4,2048,512]
  const float* ctx = (const float*)d_in[1];   // [4,2048,512]
  const float* Wq  = (const float*)d_in[2];   // [512,512]
  const float* Wk  = (const float*)d_in[3];
  const float* Wv  = (const float*)d_in[4];
  const float* Wo  = (const float*)d_in[5];
  const float* bo  = (const float*)d_in[6];   // [512]
  float* Y = (float*)d_out;

  char* ws = (char*)d_ws;
  size_t off = 0;
  auto alloc = [&](size_t bytes) -> void* {
    void* p = ws + off;
    off = (off + bytes + 255) & ~(size_t)255;
    return p;
  };
  _Float16* WqT = (_Float16*)alloc((size_t)512 * 512 * 2);
  _Float16* WkT = (_Float16*)alloc((size_t)512 * 512 * 2);
  _Float16* WvT = (_Float16*)alloc((size_t)512 * 512 * 2);
  _Float16* WoT = (_Float16*)alloc((size_t)512 * 512 * 2);
  _Float16* Qf  = (_Float16*)alloc((size_t)4 * 2048 * 512 * 2);
  _Float16* Kf  = (_Float16*)alloc((size_t)4 * 2048 * 512 * 2);
  _Float16* Vt  = (_Float16*)alloc((size_t)4 * 2048 * 512 * 2);
  _Float16* Ofb = (_Float16*)alloc((size_t)4 * 2048 * 512 * 2);

  // 0) weight transpose + f16 convert
  transpose_w_kernel<<<1024, 256, 0, stream>>>(Wq, WqT);
  transpose_w_kernel<<<1024, 256, 0, stream>>>(Wk, WkT);
  transpose_w_kernel<<<1024, 256, 0, stream>>>(Wv, WvT);
  transpose_w_kernel<<<1024, 256, 0, stream>>>(Wo, WoT);

  // 1) projections
  dim3 gProj(64, 8);
  proj_gemm_kernel<0><<<gProj, 256, 0, stream>>>(x,   WqT, Qf);
  proj_gemm_kernel<0><<<gProj, 256, 0, stream>>>(ctx, WkT, Kf);
  proj_gemm_kernel<1><<<gProj, 256, 0, stream>>>(ctx, WvT, Vt);

  // 2) flash attention
  attn_kernel<<<dim3(16, 8, 4), 256, 0, stream>>>(Qf, Kf, Vt, Ofb);

  // 3) output projection
  out_gemm_kernel<<<gProj, 256, 0, stream>>>(Ofb, WoT, bo, Y);
}
